// MemTransformerLM_47639777247729
// MI455X (gfx1250) — matrix operand order repaired and checked by reading, not compiled, not run on hardware
//
#include <hip/hip_runtime.h>
#include <math.h>

// ---------------------------------------------------------------------------
// Transformer-XL forward for gfx1250 (MI455X).  All GEMM-shaped einsums run
// through V_WMMA_F32_16X16X32_BF16 (bf16 A/B, fp32 accumulate).  B tiles are
// staged memory->LDS with GLOBAL_LOAD_ASYNC_TO_LDS_B128 (ASYNCcnt) when the
// toolchain exposes the builtin; otherwise load+ds_store fallback.
// Workspace requirement: ~893 MB.
// ---------------------------------------------------------------------------

typedef unsigned short bhalf;                                    // bf16 bits
typedef __attribute__((ext_vector_type(4)))  unsigned int u32x4;
typedef __attribute__((ext_vector_type(4)))  int          i32x4;
typedef __attribute__((ext_vector_type(8)))  float        v8f;
typedef __attribute__((ext_vector_type(16))) __bf16       v16bf;

typedef __attribute__((address_space(1))) i32x4 gas_i32x4;   // global int4
typedef __attribute__((address_space(3))) i32x4 las_i32x4;   // LDS int4

#define L_    12
#define H_    8
#define D_    512
#define DH_   64
#define DI_   2048
#define V_    10000
#define QLEN_ 1024
#define MLEN_ 1024
#define KLEN_ 2048
#define BSZ_  4
#define NTQ_  (QLEN_*BSZ_)   // 4096 query tokens
#define NTK_  (KLEN_*BSZ_)   // 8192 key tokens

#if __has_builtin(__builtin_amdgcn_global_load_async_to_lds_b128)
#define HAVE_ASYNC_LDS 1
#else
#define HAVE_ASYNC_LDS 0
#endif

__device__ __forceinline__ bhalf f2bf(float f) {
  union { float f; unsigned u; } x; x.f = f;
  unsigned r = x.u + 0x7FFFu + ((x.u >> 16) & 1u);   // round-to-nearest-even
  return (bhalf)(r >> 16);
}

__device__ __forceinline__ v16bf make_frag(u32x4 lo, u32x4 hi) {
  union { u32x4 u[2]; v16bf v; } x;
  x.u[0] = lo; x.u[1] = hi;
  return x.v;
}

// ---------------------------------------------------------------------------
// Generic WMMA GEMM:  C[M,N] (+)= A[M,K] * B[N,K]^T     (both K-contiguous)
// Block: 256 threads = 8 waves. Block tile 128(M) x 64(N), K-step 32.
// Each wave owns a 16x64 strip (4 WMMA accumulators).
// B tile (64x32 bf16) is staged through LDS and shared by all 8 waves.
// Batched via blockIdx.z: Aoff = z*sA, Boff = (z%bMod)*sB, Coff = z*sC.
// ---------------------------------------------------------------------------
__global__ void __launch_bounds__(256)
k_gemm(const bhalf* __restrict__ A, int lda, long long sA,
       const bhalf* __restrict__ B, int ldb, long long sB, int bMod,
       float* __restrict__ Cf, bhalf* __restrict__ Cbf, int ldc, long long sC,
       const float* __restrict__ bias,
       int M, int N, int K, int relu)
{
  __shared__ bhalf Bs[64][40];          // 64 rows x 32 K, padded to 40 (80B)

  int bz = blockIdx.z;
  A += (long long)bz * sA;
  B += (long long)(bz % bMod) * sB;
  long long cbase = (long long)bz * sC;

  int tid  = threadIdx.x;
  int lane = tid & 31;
  int wave = tid >> 5;
  int m0 = blockIdx.y * 128 + wave * 16;
  int n0 = blockIdx.x * 64;

  int arow = lane & 15;
  int hi   = lane >> 4;                 // 0 or 1
  int akb  = hi << 3;                   // A lane K base: 0 or 8
  int bkb  = hi << 3;                   // B lane K base: 0 or 8

  v8f acc[4] = {};

  const bhalf* Ap = A + (long long)(m0 + arow) * lda + akb;

  int ldr  = tid >> 2;                  // 0..63 : B row this thread loads
  int lseg = (tid & 3) << 3;            // 0,8,16,24
  int gn = n0 + ldr;
  if (gn >= N) gn = N - 1;              // clamp: junk only lands in discarded
  const bhalf* Bp = B + (long long)gn * ldb + lseg;   // columns (n >= N)

#if HAVE_ASYNC_LDS
  las_i32x4* ldst = (las_i32x4*)(&Bs[ldr][lseg]);
#endif

  for (int k0 = 0; k0 < K; k0 += 32) {
#if HAVE_ASYNC_LDS
    __builtin_amdgcn_global_load_async_to_lds_b128(
        (gas_i32x4*)(Bp + k0), ldst, 0, 0);
#if __has_builtin(__builtin_amdgcn_s_wait_asynccnt)
    __builtin_amdgcn_s_wait_asynccnt(0);
#else
    asm volatile("s_wait_asynccnt 0x0" ::: "memory");
#endif
    __syncthreads();
#else
    u32x4 bv = *(const u32x4*)(Bp + k0);
    *(u32x4*)(&Bs[ldr][lseg]) = bv;
    __syncthreads();
#endif

    if (k0 + 32 < K) __builtin_prefetch(Ap + k0 + 32, 0, 3);

    u32x4 alo = *(const u32x4*)(Ap + k0);
    u32x4 ahi = *(const u32x4*)(Ap + k0 + 16);
    v16bf af = make_frag(alo, ahi);

#pragma unroll
    for (int t = 0; t < 4; ++t) {
      const bhalf* bp = &Bs[t * 16 + arow][bkb];
      u32x4 blo = *(const u32x4*)(bp);
      u32x4 bhi = *(const u32x4*)(bp + 16);
      v16bf bfv = make_frag(blo, bhi);
      acc[t] = __builtin_amdgcn_wmma_f32_16x16x32_bf16(
          false, af, false, bfv, (short)0, acc[t], false, false);
    }
    __syncthreads();
  }

  // Epilogue: lane holds col n = n0+t*16+arow, rows m0 + 8*hi + v (v=0..7)
  int mbase = m0 + (hi << 3);
#pragma unroll
  for (int t = 0; t < 4; ++t) {
    int n = n0 + t * 16 + arow;
    if (n < N) {
      float bvv = bias ? bias[n] : 0.0f;
#pragma unroll
      for (int v = 0; v < 8; ++v) {
        float val = acc[t][v] + bvv;
        if (relu) val = fmaxf(val, 0.0f);
        long long off = cbase + (long long)(mbase + v) * ldc + n;
        if (Cf)  Cf[off]  = val;
        if (Cbf) Cbf[off] = f2bf(val);
      }
    }
  }
}

// ---------------------------------------------------------------------------
// Elementwise / helper kernels
// ---------------------------------------------------------------------------
__global__ void __launch_bounds__(256)
k_cvt_bf16(const float* __restrict__ src, bhalf* __restrict__ dst, long long n)
{
  long long i = (long long)blockIdx.x * 256 + threadIdx.x;
  if (i < n) dst[i] = f2bf(src[i]);
}

__global__ void __launch_bounds__(256)
k_embed(const int* __restrict__ x, const float* __restrict__ emb,
        float* __restrict__ h, bhalf* __restrict__ hbf)
{
  long long idx = (long long)blockIdx.x * 256 + threadIdx.x;   // NTQ_*D_
  int d = (int)(idx & (D_ - 1));
  long long t = idx >> 9;
  int tok = x[t];
  float v = emb[(long long)tok * D_ + d] * 22.62741699796952f;  // sqrt(512)
  h[idx]   = v;
  hbf[idx] = f2bf(v);
}

__global__ void __launch_bounds__(256)
k_pos(bhalf* __restrict__ r)                                   // r[KLEN_][D_]
{
  int idx = blockIdx.x * 256 + threadIdx.x;                    // KLEN_*256
  int dp = idx & 255;
  int j  = idx >> 8;
  float pos  = (float)(KLEN_ - 1 - j);
  float invf = __expf(-9.210340371976184f * ((float)(2 * dp) / 512.0f));
  float a = pos * invf;
  r[(long long)j * D_ + dp]       = f2bf(__sinf(a));
  r[(long long)j * D_ + 256 + dp] = f2bf(__cosf(a));
}

__global__ void __launch_bounds__(256)
k_cat(const float* __restrict__ memsL, const bhalf* __restrict__ hbf,
      bhalf* __restrict__ cat)
{
  long long idx = (long long)blockIdx.x * 256 + threadIdx.x;   // NTK_*D_
  int d = (int)(idx & (D_ - 1));
  long long rowk = idx >> 9;          // j*BSZ_+b
  int b = (int)(rowk & 3);
  long long j = rowk >> 2;
  if (j < MLEN_)
    cat[idx] = f2bf(memsL[(j * BSZ_ + b) * D_ + d]);
  else
    cat[idx] = hbf[((j - MLEN_) * BSZ_ + b) * D_ + d];
}

// split heads[NTK_,3*512] -> (q+rwb)bf, (q+rrb)bf, k bf, v^T bf
__global__ void __launch_bounds__(256)
k_split(const float* __restrict__ heads,
        const float* __restrict__ rwb, const float* __restrict__ rrb,
        bhalf* __restrict__ qw, bhalf* __restrict__ qr,
        bhalf* __restrict__ kk, bhalf* __restrict__ vt)
{
  long long idx = (long long)blockIdx.x * 256 + threadIdx.x;   // NTK_*D_
  int c = (int)(idx & (D_ - 1));
  long long rowk = idx >> 9;
  int b = (int)(rowk & 3);
  long long j = rowk >> 2;
  const float* hrow = heads + rowk * (3 * D_);

  kk[idx] = f2bf(hrow[D_ + c]);

  float vv = hrow[2 * D_ + c];
  int n = c >> 6, d = c & 63;
  vt[((long long)(b * H_ + n) * DH_ + d) * KLEN_ + j] = f2bf(vv);

  if (j >= MLEN_) {
    long long i = j - MLEN_;
    float qv = hrow[c];
    long long qidx = (i * BSZ_ + b) * D_ + c;
    qw[qidx] = f2bf(qv + rwb[c]);
    qr[qidx] = f2bf(qv + rrb[c]);
  }
}

// S[bn][i][j] = (S + rel_shift(P))[.] * scale,  masked (j > i+MLEN) -> -inf
__global__ void __launch_bounds__(256)
k_shift_mask(float* __restrict__ S, const float* __restrict__ Pf)
{
  long long idx = (long long)blockIdx.x * 256 + threadIdx.x;   // 32*Q*K
  int j = (int)(idx & (KLEN_ - 1));
  int i = (int)((idx >> 11) & (QLEN_ - 1));
  long long bn = idx >> 21;
  int u = j - i + (QLEN_ - 1);
  float bd = 0.0f;
  if (u >= 0 && u < KLEN_)
    bd = Pf[(bn << 21) + ((long long)i << 11) + u];
  float v = (S[idx] + bd) * 0.125f;                            // 1/sqrt(64)
  if (j > i + MLEN_) v = -INFINITY;
  S[idx] = v;
}

__global__ void __launch_bounds__(256)
k_softmax(const float* __restrict__ S, bhalf* __restrict__ P)
{
  __shared__ float red[256];
  long long base = (long long)blockIdx.x * KLEN_;
  int tid = threadIdx.x;
  float m = -3.4e38f;
  for (int j = tid; j < KLEN_; j += 256) m = fmaxf(m, S[base + j]);
  red[tid] = m; __syncthreads();
  for (int s = 128; s > 0; s >>= 1) {
    if (tid < s) red[tid] = fmaxf(red[tid], red[tid + s]);
    __syncthreads();
  }
  m = red[0]; __syncthreads();
  float sum = 0.0f;
  for (int j = tid; j < KLEN_; j += 256) sum += __expf(S[base + j] - m);
  red[tid] = sum; __syncthreads();
  for (int s = 128; s > 0; s >>= 1) {
    if (tid < s) red[tid] += red[tid + s];
    __syncthreads();
  }
  float inv = 1.0f / red[0];
  for (int j = tid; j < KLEN_; j += 256)
    P[base + j] = f2bf(__expf(S[base + j] - m) * inv);
}

// h = LayerNorm(h + add) * g + b ; also refresh bf16 shadow
__global__ void __launch_bounds__(256)
k_add_ln(float* __restrict__ h, const float* __restrict__ add,
         const float* __restrict__ g, const float* __restrict__ bb,
         bhalf* __restrict__ hbf)
{
  __shared__ float red[256];
  int tid = threadIdx.x;
  long long base = (long long)blockIdx.x * D_;
  float x0 = h[base + tid]       + add[base + tid];
  float x1 = h[base + 256 + tid] + add[base + 256 + tid];
  red[tid] = x0 + x1; __syncthreads();
  for (int s = 128; s > 0; s >>= 1) {
    if (tid < s) red[tid] += red[tid + s];
    __syncthreads();
  }
  float mean = red[0] * (1.0f / 512.0f);
  __syncthreads();
  float d0 = x0 - mean, d1 = x1 - mean;
  red[tid] = d0 * d0 + d1 * d1; __syncthreads();
  for (int s = 128; s > 0; s >>= 1) {
    if (tid < s) red[tid] += red[tid + s];
    __syncthreads();
  }
  float rstd = rsqrtf(red[0] * (1.0f / 512.0f) + 1e-5f);
  float y0 = d0 * rstd * g[tid]       + bb[tid];
  float y1 = d1 * rstd * g[256 + tid] + bb[256 + tid];
  h[base + tid]         = y0;
  h[base + 256 + tid]   = y1;
  hbf[base + tid]       = f2bf(y0);
  hbf[base + 256 + tid] = f2bf(y1);
}

// ---------------------------------------------------------------------------
// Host orchestration
// ---------------------------------------------------------------------------
extern "C" void kernel_launch(void* const* d_in, const int* in_sizes, int n_in,
                              void* d_out, int out_size, void* d_ws, size_t ws_size,
                              hipStream_t stream)
{
  (void)in_sizes; (void)n_in; (void)out_size; (void)ws_size;

  const int*   x      = (const int*)  d_in[0];
  const float* mems   = (const float*)d_in[1];
  const float* emb    = (const float*)d_in[2];
  const float* qkv_w  = (const float*)d_in[3];
  const float* rnet_w = (const float*)d_in[4];
  const float* o_w    = (const float*)d_in[5];
  const float* ln1_g  = (const float*)d_in[6];
  const float* ln1_b  = (const float*)d_in[7];
  const float* ff_w1  = (const float*)d_in[8];
  const float* ff_b1  = (const float*)d_in[9];
  const float* ff_w2  = (const float*)d_in[10];
  const float* ff_b2  = (const float*)d_in[11];
  const float* ln2_g  = (const float*)d_in[12];
  const float* ln2_b  = (const float*)d_in[13];
  const float* rwb    = (const float*)d_in[14];
  const float* rrb    = (const float*)d_in[15];
  const float* proj_w = (const float*)d_in[16];
  const float* proj_b = (const float*)d_in[17];
  float* logits = (float*)d_out;

  char* p = (char*)d_ws;
  auto alloc = [&](size_t bytes) -> void* {
    void* r = (void*)p;
    p += (bytes + 255) & ~(size_t)255;
    return r;
  };

  // bf16 weights
  bhalf* wqkv  = (bhalf*)alloc((size_t)L_ * 3 * D_ * D_ * 2);
  bhalf* wrnet = (bhalf*)alloc((size_t)L_ * D_ * D_ * 2);
  bhalf* wow   = (bhalf*)alloc((size_t)L_ * D_ * D_ * 2);
  bhalf* wff1  = (bhalf*)alloc((size_t)L_ * DI_ * D_ * 2);
  bhalf* wff2  = (bhalf*)alloc((size_t)L_ * D_ * DI_ * 2);
  bhalf* wproj = (bhalf*)alloc((size_t)V_ * D_ * 2);
  // activations
  float* h     = (float*)alloc((size_t)NTQ_ * D_ * 4);
  bhalf* hbf   = (bhalf*)alloc((size_t)NTQ_ * D_ * 2);
  bhalf* rbf   = (bhalf*)alloc((size_t)KLEN_ * D_ * 2);
  bhalf* cat   = (bhalf*)alloc((size_t)NTK_ * D_ * 2);
  float* heads = (float*)alloc((size_t)NTK_ * 3 * D_ * 4);
  bhalf* qwb   = (bhalf*)alloc((size_t)NTQ_ * D_ * 2);
  bhalf* qrb   = (bhalf*)alloc((size_t)NTQ_ * D_ * 2);
  bhalf* kbf   = (bhalf*)alloc((size_t)NTK_ * D_ * 2);
  bhalf* vt    = (bhalf*)alloc((size_t)BSZ_ * H_ * DH_ * KLEN_ * 2);
  bhalf* rk    = (bhalf*)alloc((size_t)KLEN_ * D_ * 2);
  float* S     = (float*)alloc((size_t)BSZ_ * H_ * QLEN_ * KLEN_ * 4);
  float* Pf    = (float*)alloc((size_t)BSZ_ * H_ * QLEN_ * KLEN_ * 4);
  bhalf* probs = (bhalf*)alloc((size_t)BSZ_ * H_ * QLEN_ * KLEN_ * 2);
  bhalf* avbf  = (bhalf*)alloc((size_t)NTQ_ * D_ * 2);
  float* tmp   = (float*)alloc((size_t)NTQ_ * D_ * 4);
  bhalf* ff1bf = (bhalf*)alloc((size_t)NTQ_ * DI_ * 2);

  const int BIG = 1 << 30;

  auto cvt = [&](const float* src, bhalf* dst, long long n) {
    k_cvt_bf16<<<dim3((unsigned)((n + 255) / 256)), 256, 0, stream>>>(src, dst, n);
  };
  auto gemm = [&](const bhalf* A, int lda, long long sA,
                  const bhalf* B, int ldb, long long sB, int bMod,
                  float* Cf, bhalf* Cbf, int ldc, long long sC,
                  const float* bias, int M, int N, int K, int relu, int batches) {
    dim3 g((unsigned)((N + 63) / 64), (unsigned)(M / 128), (unsigned)batches);
    k_gemm<<<g, 256, 0, stream>>>(A, lda, sA, B, ldb, sB, bMod,
                                  Cf, Cbf, ldc, sC, bias, M, N, K, relu);
  };

  // 1) weights -> bf16
  cvt(qkv_w,  wqkv,  (long long)L_ * 3 * D_ * D_);
  cvt(rnet_w, wrnet, (long long)L_ * D_ * D_);
  cvt(o_w,    wow,   (long long)L_ * D_ * D_);
  cvt(ff_w1,  wff1,  (long long)L_ * DI_ * D_);
  cvt(ff_w2,  wff2,  (long long)L_ * D_ * DI_);
  cvt(proj_w, wproj, (long long)V_ * D_);

  // 2) embedding + positional encodings
  k_embed<<<dim3((NTQ_ * D_) / 256), 256, 0, stream>>>(x, emb, h, hbf);
  k_pos<<<dim3((KLEN_ * 256) / 256), 256, 0, stream>>>(rbf);

  const long long QK = (long long)QLEN_ * KLEN_;   // 2^21

  for (int l = 0; l < L_; ++l) {
    const float* memsL = mems + (size_t)l * MLEN_ * BSZ_ * D_;

    // cat = [mems_l ; h]  (bf16)
    k_cat<<<dim3((NTK_ * D_) / 256), 256, 0, stream>>>(memsL, hbf, cat);

    // heads = cat @ qkv_w^T    [8192 x 1536]
    gemm(cat, D_, 0, wqkv + (size_t)l * 3 * D_ * D_, D_, 0, BIG,
         heads, nullptr, 3 * D_, 0, nullptr, NTK_, 3 * D_, D_, 0, 1);

    // split into q(+biases), k, v^T
    k_split<<<dim3((NTK_ * D_) / 256), 256, 0, stream>>>(heads, rwb, rrb,
                                                         qwb, qrb, kbf, vt);

    // r_head_k = r @ r_net_w^T   [2048 x 512] -> bf16 [u][n][dh]
    gemm(rbf, D_, 0, wrnet + (size_t)l * D_ * D_, D_, 0, BIG,
         nullptr, rk, D_, 0, nullptr, KLEN_, D_, D_, 0, 1);

    // AC[bn][i][j] = (q+rwb) . k     batched over 32 (b,h)
    gemm(qwb, BSZ_ * D_, 64, kbf, BSZ_ * D_, 64, BIG,
         S, nullptr, KLEN_, QK, nullptr, QLEN_, KLEN_, DH_, 0, BSZ_ * H_);

    // P[bn][i][u] = (q+rrb) . r_head_k   (B batched by head only)
    gemm(qrb, BSZ_ * D_, 64, rk, H_ * DH_, 64, H_,
         Pf, nullptr, KLEN_, QK, nullptr, QLEN_, KLEN_, DH_, 0, BSZ_ * H_);

    // S = (AC + rel_shift(P)) * scale, mask
    k_shift_mask<<<dim3((unsigned)((32 * QK) / 256)), 256, 0, stream>>>(S, Pf);

    // softmax over j -> probs (bf16)
    k_softmax<<<dim3(BSZ_ * H_ * QLEN_), 256, 0, stream>>>(S, probs);

    // attn_vec[i][b][h*dh] = probs @ v   (batched, K=2048, N=64)
    gemm(probs, KLEN_, QK, vt, KLEN_, (long long)DH_ * KLEN_, BIG,
         nullptr, avbf, BSZ_ * D_, 64, nullptr, QLEN_, DH_, KLEN_, 0, BSZ_ * H_);

    // attn_out = attn_vec @ o_w^T
    gemm(avbf, D_, 0, wow + (size_t)l * D_ * D_, D_, 0, BIG,
         tmp, nullptr, D_, 0, nullptr, NTQ_, D_, D_, 0, 1);

    // h = LN(h + attn_out)
    k_add_ln<<<dim3(NTQ_), 256, 0, stream>>>(h, tmp,
                                             ln1_g + (size_t)l * D_,
                                             ln1_b + (size_t)l * D_, hbf);

    // ff1 = relu(h @ w1^T + b1)  (bf16)
    gemm(hbf, D_, 0, wff1 + (size_t)l * DI_ * D_, D_, 0, BIG,
         nullptr, ff1bf, DI_, 0, ff_b1 + (size_t)l * DI_, NTQ_, DI_, D_, 1, 1);

    // ff2 = ff1 @ w2^T + b2
    gemm(ff1bf, DI_, 0, wff2 + (size_t)l * D_ * DI_, DI_, 0, BIG,
         tmp, nullptr, D_, 0, ff_b2 + (size_t)l * D_, NTQ_, D_, DI_, 0, 1);

    // h = LN(h + ff2)
    k_add_ln<<<dim3(NTQ_), 256, 0, stream>>>(h, tmp,
                                             ln2_g + (size_t)l * D_,
                                             ln2_b + (size_t)l * D_, hbf);
  }

  // logits = h @ proj_w^T + proj_b   [4096 x 10000]
  gemm(hbf, D_, 0, wproj, D_, 0, BIG,
       logits, nullptr, V_, 0, proj_b, NTQ_, V_, D_, 0, 1);
}
